// RNNModel_11699490914552
// MI455X (gfx1250) — matrix operand order repaired and checked by reading, not compiled, not run on hardware
//
#include <hip/hip_runtime.h>
#include <hip/hip_bf16.h>
#include <cstdint>

// ---------------------------------------------------------------------------
// Problem constants
// ---------------------------------------------------------------------------
#define BB   256      // batch
#define LL   70       // seq len
#define CLn  16       // char len
#define CEn  50       // char emb
#define WDn  300      // word emb
#define FDn  20       // feat dim
#define HH   150      // hidden
#define FF   100      // conv filters
#define NCn  5        // classes
#define NG   608      // 4H=600 padded to 608 (38 n-tiles of 16)
#define HK   160      // H padded to 160 (5 k-chunks of 32)
#define K0P  448      // layer-0 input dim 420 padded to 448
#define KLP  320      // layer-1..4 input dim 300 padded to 320
#define MR   (BB*LL)  // 17920 rows

typedef __attribute__((ext_vector_type(16))) _Float16 v16h;
typedef __attribute__((ext_vector_type(8)))  _Float16 v8h;
typedef __attribute__((ext_vector_type(8)))  float    v8f;

// ---------------------------------------------------------------------------
// Workspace layout (bytes). embPad aliases xW (disjoint lifetimes).
// ---------------------------------------------------------------------------
static constexpr size_t OFF_LENS   = 0;                                 // 256*4
static constexpr size_t OFF_CONVWT = 1024;                              // 112*5*64*2 = 71680
static constexpr size_t OFF_WX0    = 73728;                             // 2 * 608*448*2 = 1089536
static constexpr size_t OFF_WH0    = OFF_WX0 + 1089536;                 // 2 * 608*160*2 = 389120
static constexpr size_t OFF_WXL    = OFF_WH0 + 389120;                  // 8 * 608*320*2 = 3112960
static constexpr size_t OFF_WHL    = OFF_WXL + 3112960;                 // 8 * 608*160*2 = 1556480
static constexpr size_t OFF_BIAS   = OFF_WHL + 1556480;                 // 10 * 608*4   = 24320
static constexpr size_t OFF_BUF0   = 6291456;                           // 17920*448*2 = 16056320
static constexpr size_t OFF_BUF1   = OFF_BUF0 + 16056320;               // 17920*320*2 = 11468800
static constexpr size_t OFF_BUF2   = OFF_BUF1 + 11468800;               // 11468800
static constexpr size_t OFF_Z      = OFF_BUF2 + 11468800;               // 2*256*608*4 = 1245184
static constexpr size_t OFF_BIG    = OFF_Z + 1245184;                   // max(embPad 45.9MB, xW 87.2MB)
static constexpr size_t TOTAL_WS   = OFF_BIG + (size_t)2*MR*NG*4;       // ~133.7 MB

// ---------------------------------------------------------------------------
// Helpers
// ---------------------------------------------------------------------------
static __device__ __forceinline__ v16h cat8(v8h a, v8h b) {
  return __builtin_shufflevector(a, b, 0,1,2,3,4,5,6,7,8,9,10,11,12,13,14,15);
}
static __device__ __forceinline__ v8f wmma16(v16h a, v16h b, v8f c) {
  // D = A(16x32 f16) * B(32x16 f16) + C(16x16 f32)
  return __builtin_amdgcn_wmma_f32_16x16x32_f16(false, a, false, b, (short)0, c, false, false);
}
static __device__ __forceinline__ float sigf(float x) {
  return 1.0f / (1.0f + __expf(-x));
}

// ---------------------------------------------------------------------------
// Small prep kernels
// ---------------------------------------------------------------------------
__global__ void zero_ws_k(uint4* p, size_t n16) {
  size_t i = (size_t)blockIdx.x * blockDim.x + threadIdx.x;
  size_t st = (size_t)gridDim.x * blockDim.x;
  uint4 z; z.x = z.y = z.z = z.w = 0u;
  for (; i < n16; i += st) p[i] = z;
}

__global__ void lens_k(const int* __restrict__ mask, int* __restrict__ lens) {
  int b = threadIdx.x;
  if (b < BB) {
    int s = 0;
    for (int t = 0; t < LL; ++t) s += mask[b * LL + t];
    lens[b] = s;
  }
}

__global__ void fill_x0_k(const float* __restrict__ X, const float* __restrict__ feat,
                          _Float16* __restrict__ x0) {
  int idx = blockIdx.x * 256 + threadIdx.x;
  if (idx >= MR * (WDn + FDn)) return;
  int row = idx / (WDn + FDn), col = idx % (WDn + FDn);
  float v; int c2;
  if (col < WDn) { v = X[(size_t)row * WDn + col]; c2 = col; }
  else           { v = feat[(size_t)row * FDn + (col - WDn)]; c2 = WDn + FF + (col - WDn); }
  x0[(size_t)row * K0P + c2] = (_Float16)v;
}

__global__ void fill_emb_k(const float* __restrict__ Xc, _Float16* __restrict__ embPad) {
  int idx = blockIdx.x * 256 + threadIdx.x;
  if (idx >= MR * CLn * CEn) return;
  int w = idx / (CLn * CEn), r = idx % (CLn * CEn);
  int p = r / CEn, c = r % CEn;
  embPad[(size_t)w * (20 * 64) + (size_t)(p + 2) * 64 + c] = (_Float16)Xc[idx];
}

__global__ void conv_wt_k(const float* __restrict__ ck, _Float16* __restrict__ dst) {
  int idx = blockIdx.x * 256 + threadIdx.x;
  if (idx >= 5 * CEn * FF) return;
  int k = idx / (CEn * FF), c = (idx / FF) % CEn, f = idx % FF;
  dst[(size_t)f * (5 * 64) + (size_t)k * 64 + c] = (_Float16)ck[idx];
}

// dst[n][k] = src[k][n], dst stride Kpad (pre-zeroed padding)
__global__ void transpose_w_k(const float* __restrict__ src, _Float16* __restrict__ dst,
                              int Krows, int Kpad) {
  int idx = blockIdx.x * 256 + threadIdx.x;
  if (idx >= Krows * 600) return;
  int k = idx / 600, n = idx % 600;
  dst[(size_t)n * Kpad + k] = (_Float16)src[idx];
}

__global__ void copy_bias_k(const float* __restrict__ src, float* __restrict__ dst) {
  int idx = blockIdx.x * 256 + threadIdx.x;
  if (idx < 600) dst[idx] = src[idx];
}

// ---------------------------------------------------------------------------
// Char CNN via WMMA: conv1d(SAME,K=5) + bias + max over 16 char positions + relu,
// fused into the WMMA epilogue. One 16x16 tile == one word (all 16 char positions).
// ---------------------------------------------------------------------------
__global__ __launch_bounds__(256) void char_conv_k(
    const _Float16* __restrict__ embPad,   // [MR][20][64]
    const _Float16* __restrict__ cwt,      // [112][5][64]  cwt[f][tap][c]
    const float* __restrict__ convb,       // [100]
    _Float16* __restrict__ x0)             // [MR][448], writes cols 300..399
{
  int gw = blockIdx.x * 8 + (threadIdx.x >> 5);
  int lane = threadIdx.x & 31, hi = lane >> 4, lo = lane & 15;
  int word = gw / 7, nt = gw % 7;
  int f = nt * 16 + lo;
  v8f acc = {0.f,0.f,0.f,0.f,0.f,0.f,0.f,0.f};
  const _Float16* wbase = embPad + (size_t)word * (20 * 64);
  #pragma unroll
  for (int tap = 0; tap < 5; ++tap) {
    const _Float16* ap = wbase + (size_t)(lo + tap) * 64;        // A row m=lo (char pos)
    const _Float16* bp = cwt + (size_t)f * (5 * 64) + tap * 64;  // B col n=f
    #pragma unroll
    for (int kk = 0; kk < 2; ++kk) {
      int kb = kk * 32;
      v8h aLo = *(const v8h*)(ap + kb + hi * 8);
      v8h aHi = *(const v8h*)(ap + kb + 16 + hi * 8);
      v8h bLo = *(const v8h*)(bp + kb + hi * 8);
      v8h bHi = *(const v8h*)(bp + kb + 16 + hi * 8);
      acc = wmma16(cat8(aLo, aHi), cat8(bLo, bHi), acc);
    }
  }
  float mx = acc[0];
  #pragma unroll
  for (int r = 1; r < 8; ++r) mx = fmaxf(mx, acc[r]);
  float other = __shfl_xor(mx, 16, 32);       // combine the two 8-row halves
  mx = fmaxf(mx, other);
  float cb = (f < FF) ? convb[f] : 0.f;
  float val = fmaxf(mx + cb, 0.f);            // max_p relu(conv+b) == relu(b + max_p conv)
  if (hi == 0 && f < FF)
    x0[(size_t)word * K0P + WDn + f] = (_Float16)val;
}

// ---------------------------------------------------------------------------
// Input projection GEMM: Y[M=17920][608] = Xm[M][Kpad] * Wt^T + bias
// One wave = one 16x16 tile. Fragments load straight from global (L2-resident).
// ---------------------------------------------------------------------------
__global__ __launch_bounds__(256) void gemm_xw_k(
    const _Float16* __restrict__ Xm,  // [MR][Kpad]
    const _Float16* __restrict__ Wt,  // [608][Kpad]  Wt[n][k] = W[k][n]
    const float* __restrict__ bias,   // [608]
    float* __restrict__ Y,            // [MR][608]
    int Kpad, int kIters)
{
  int gw = blockIdx.x * 8 + (threadIdx.x >> 5);
  int lane = threadIdx.x & 31, hi = lane >> 4, lo = lane & 15;
  int m0 = (gw / 38) * 16, n0 = (gw % 38) * 16;
  int n = n0 + lo;
  const _Float16* ap = Xm + (size_t)(m0 + lo) * Kpad;
  const _Float16* bp = Wt + (size_t)n * Kpad;
  v8f acc = {0.f,0.f,0.f,0.f,0.f,0.f,0.f,0.f};
  for (int kk = 0; kk < kIters; ++kk) {
    int kb = kk * 32;
    __builtin_prefetch(ap + kb + 64, 0, 1);
    __builtin_prefetch(bp + kb + 64, 0, 1);
    v8h aLo = *(const v8h*)(ap + kb + hi * 8);
    v8h aHi = *(const v8h*)(ap + kb + 16 + hi * 8);
    v8h bLo = *(const v8h*)(bp + kb + hi * 8);
    v8h bHi = *(const v8h*)(bp + kb + 16 + hi * 8);
    acc = wmma16(cat8(aLo, aHi), cat8(bLo, bHi), acc);
  }
  float bv = bias[n];
  #pragma unroll
  for (int r = 0; r < 8; ++r)
    Y[(size_t)(m0 + hi * 8 + r) * NG + n] = acc[r] + bv;
}

// ---------------------------------------------------------------------------
// Recurrent scan: one workgroup per direction (grid=2), 1024 threads = 32 waves.
// Per step: z = gather(xW) + h @ Wh (WMMA, h in LDS f16), then gates (c in regs),
// with dynamic_rnn masking + per-example time reversal for the bw direction.
// ---------------------------------------------------------------------------
__global__ __launch_bounds__(1024) void lstm_layer_k(
    const _Float16* __restrict__ WhT,  // dir0 at base; dir1 at +whStride halves
    const float* __restrict__ xW,      // dir0 at base; dir1 at +xwStride floats
    const int* __restrict__ lens,
    _Float16* __restrict__ outBuf,     // [MR][320], fw cols 0..149, bw 150..299
    float* __restrict__ zbuf)          // [2][256][608]
{
  const int dir = blockIdx.x;
  const _Float16* Wh = WhT + (size_t)dir * (NG * HK);
  const float* xw = xW + (size_t)dir * ((size_t)MR * NG);
  float* zb = zbuf + (size_t)dir * (BB * NG);

  __shared__ _Float16 hS[BB * HK];
  __shared__ int lenS[BB];
  int tid = threadIdx.x;
  for (int i = tid; i < BB * HK; i += 1024) hS[i] = (_Float16)0.f;
  if (tid < BB) lenS[tid] = lens[tid];
  float c[38];
  #pragma unroll
  for (int i = 0; i < 38; ++i) c[i] = 0.f;
  __syncthreads();

  int wave = tid >> 5, lane = tid & 31, hi = lane >> 4, lo = lane & 15;

  for (int s = 0; s < LL; ++s) {
    // ---- GEMM phase: 608 tiles over 32 waves (19 each)
    for (int it = 0; it < 19; ++it) {
      int tile = wave + it * 32;
      int m0 = (tile / 38) * 16, n0 = (tile % 38) * 16;
      int n = n0 + lo;
      v8f acc;
      #pragma unroll
      for (int r = 0; r < 8; ++r) {           // seed accumulator with gathered xW
        int b = m0 + hi * 8 + r;
        int len = lenS[b];
        int tix = (dir == 0) ? s : ((s < len) ? (len - 1 - s) : s);
        acc[r] = xw[(size_t)(b * LL + tix) * NG + n];
      }
      const _Float16* ap = hS + (size_t)(m0 + lo) * HK + hi * 8;
      const _Float16* bp = Wh + (size_t)n * HK;
      #pragma unroll
      for (int kk = 0; kk < 5; ++kk) {        // K = 160 (150 + zero pad)
        int kb = kk * 32;
        v8h aLo = *(const v8h*)(ap + kb);
        v8h aHi = *(const v8h*)(ap + kb + 16);
        v8h bLo = *(const v8h*)(bp + kb + hi * 8);
        v8h bHi = *(const v8h*)(bp + kb + 16 + hi * 8);
        acc = wmma16(cat8(aLo, aHi), cat8(bLo, bHi), acc);
      }
      #pragma unroll
      for (int r = 0; r < 8; ++r)
        zb[(size_t)(m0 + hi * 8 + r) * NG + n] = acc[r];
    }
    asm volatile("s_wait_storecnt 0x0" ::: "memory");  // drain this wave's z stores
    __threadfence();                                   // make z visible across the WGP
    __syncthreads();

    // ---- Gate phase: 38400 cells, c in registers (full unroll keeps VGPRs)
    #pragma unroll
    for (int i = 0; i < 38; ++i) {
      int cell = tid + i * 1024;
      if (cell < BB * HH) {
        int b = cell / HH, n = cell % HH;
        int len = lenS[b];
        bool m = (s < len);
        float zi = zb[b * NG + n];
        float zj = zb[b * NG + n + HH];
        float zf = zb[b * NG + n + 2 * HH];
        float zo = zb[b * NG + n + 3 * HH];
        float cn = sigf(zf + 1.0f) * c[i] + sigf(zi) * tanhf(zj);
        float hn = sigf(zo) * tanhf(cn);
        float ho;
        if (m) { c[i] = cn; hS[b * HK + n] = (_Float16)hn; ho = hn; }
        else   { ho = 0.f; }
        int t = (dir == 0) ? s : (m ? (len - 1 - s) : s);
        outBuf[(size_t)(b * LL + t) * KLP + dir * HH + n] = (_Float16)ho;
      }
    }
    __syncthreads();   // hS updates visible before next step's GEMM
  }
}

// ---------------------------------------------------------------------------
// Head: gather last valid timestep, FC+relu, output projection. Tiny (23 MFLOP).
// ---------------------------------------------------------------------------
__global__ __launch_bounds__(160) void final_k(
    const _Float16* __restrict__ xin,  // [MR][320] cols 0..299 used
    const float* __restrict__ fcw, const float* __restrict__ fcb,
    const float* __restrict__ ow, const float* __restrict__ ob,
    const int* __restrict__ lens, float* __restrict__ out)
{
  __shared__ float hsh[HH];
  int b = blockIdx.x, j = threadIdx.x;
  int len = lens[b];
  int t = (len > 0) ? (len - 1) : 0;
  const _Float16* xr = xin + (size_t)(b * LL + t) * KLP;
  if (j < HH) {
    float a = fcb[j];
    for (int k = 0; k < 2 * HH; ++k) a += (float)xr[k] * fcw[k * HH + j];
    hsh[j] = fmaxf(a, 0.f);
  }
  __syncthreads();
  if (j < NCn) {
    float a = ob[j];
    for (int k = 0; k < HH; ++k) a += hsh[k] * ow[k * NCn + j];
    out[b * NCn + j] = a;
  }
}

// ---------------------------------------------------------------------------
// Launch
// ---------------------------------------------------------------------------
extern "C" void kernel_launch(void* const* d_in, const int* in_sizes, int n_in,
                              void* d_out, int out_size, void* d_ws, size_t ws_size,
                              hipStream_t stream) {
  (void)in_sizes; (void)n_in; (void)out_size; (void)ws_size;
  const float* X    = (const float*)d_in[0];
  const float* Xc   = (const float*)d_in[1];
  const int*   Xm   = (const int*)d_in[2];
  const float* feat = (const float*)d_in[3];
  const float* ck   = (const float*)d_in[4];
  const float* cb   = (const float*)d_in[5];
  const float* l0fk = (const float*)d_in[6];
  const float* l0fb = (const float*)d_in[7];
  const float* l0bk = (const float*)d_in[8];
  const float* l0bb = (const float*)d_in[9];
  const float* lfk  = (const float*)d_in[10];
  const float* lfb  = (const float*)d_in[11];
  const float* lbk  = (const float*)d_in[12];
  const float* lbb  = (const float*)d_in[13];
  const float* fcw  = (const float*)d_in[14];
  const float* fcb  = (const float*)d_in[15];
  const float* oww  = (const float*)d_in[16];
  const float* owb  = (const float*)d_in[17];

  char* ws = (char*)d_ws;
  int*       lens  = (int*)(ws + OFF_LENS);
  _Float16*  cwt   = (_Float16*)(ws + OFF_CONVWT);
  _Float16*  WX0   = (_Float16*)(ws + OFF_WX0);   // 2 x [608][448]
  _Float16*  WH0   = (_Float16*)(ws + OFF_WH0);   // 2 x [608][160]
  _Float16*  WXL   = (_Float16*)(ws + OFF_WXL);   // 8 x [608][320]
  _Float16*  WHL   = (_Float16*)(ws + OFF_WHL);   // 8 x [608][160]
  float*     BIASB = (float*)(ws + OFF_BIAS);     // 10 x [608]
  _Float16*  buf0  = (_Float16*)(ws + OFF_BUF0);  // [MR][448]
  _Float16*  buf1  = (_Float16*)(ws + OFF_BUF1);  // [MR][320]
  _Float16*  buf2  = (_Float16*)(ws + OFF_BUF2);  // [MR][320]
  float*     zbuf  = (float*)(ws + OFF_Z);        // [2][256][608]
  _Float16*  embP  = (_Float16*)(ws + OFF_BIG);   // [MR][20][64] (aliases xW)
  float*     xWf   = (float*)(ws + OFF_BIG);      // [MR][608] fw
  float*     xWb   = xWf + (size_t)MR * NG;       //            bw

  // 0) deterministic scratch init (zeroes all pads, weights regions, buffers)
  zero_ws_k<<<4096, 256, 0, stream>>>((uint4*)ws, TOTAL_WS / 16);

  // 1) prep
  lens_k<<<1, 256, 0, stream>>>(Xm, lens);
  fill_x0_k<<<(MR * 320 + 255) / 256, 256, 0, stream>>>(X, feat, buf0);
  fill_emb_k<<<(MR * CLn * CEn + 255) / 256, 256, 0, stream>>>(Xc, embP);
  conv_wt_k<<<(5 * CEn * FF + 255) / 256, 256, 0, stream>>>(ck, cwt);

  auto T = [&](const float* src, _Float16* dst, int Krows, int Kpad) {
    int N = Krows * 600;
    transpose_w_k<<<(N + 255) / 256, 256, 0, stream>>>(src, dst, Krows, Kpad);
  };
  T(l0fk,                WX0,                420, K0P);
  T(l0fk + 420 * 600,    WH0,                150, HK);
  T(l0bk,                WX0 + (size_t)NG * K0P, 420, K0P);
  T(l0bk + 420 * 600,    WH0 + (size_t)NG * HK,  150, HK);
  for (int i = 0; i < 4; ++i) {
    T(lfk + (size_t)i * 450 * 600,             WXL + (size_t)(i * 2 + 0) * NG * KLP, 300, KLP);
    T(lfk + (size_t)i * 450 * 600 + 300 * 600, WHL + (size_t)(i * 2 + 0) * NG * HK,  150, HK);
    T(lbk + (size_t)i * 450 * 600,             WXL + (size_t)(i * 2 + 1) * NG * KLP, 300, KLP);
    T(lbk + (size_t)i * 450 * 600 + 300 * 600, WHL + (size_t)(i * 2 + 1) * NG * HK,  150, HK);
  }
  copy_bias_k<<<3, 256, 0, stream>>>(l0fb, BIASB + 0 * NG);
  copy_bias_k<<<3, 256, 0, stream>>>(l0bb, BIASB + 1 * NG);
  for (int i = 0; i < 4; ++i) {
    copy_bias_k<<<3, 256, 0, stream>>>(lfb + (size_t)i * 600, BIASB + (size_t)((i + 1) * 2 + 0) * NG);
    copy_bias_k<<<3, 256, 0, stream>>>(lbb + (size_t)i * 600, BIASB + (size_t)((i + 1) * 2 + 1) * NG);
  }

  // 2) char CNN (WMMA, fused bias+max+relu) -> buf0 cols 300..399
  char_conv_k<<<(MR * 7) / 8, 256, 0, stream>>>(embP, cwt, cb, buf0);

  // 3) BiLSTM stack
  const int tilesBlocks = (MR / 16) * 38 / 8;  // 5320
  for (int li = 0; li < 5; ++li) {
    const _Float16* in  = (li == 0) ? buf0 : ((li % 2 == 1) ? buf1 : buf2);
    _Float16*       out = (li % 2 == 0) ? buf1 : buf2;
    int Kpad = li ? KLP : K0P;
    const _Float16* Wxf = li ? (WXL + (size_t)((li - 1) * 2 + 0) * NG * KLP) : WX0;
    const _Float16* Wxb = li ? (WXL + (size_t)((li - 1) * 2 + 1) * NG * KLP)
                             : (WX0 + (size_t)NG * K0P);
    const _Float16* Wh0 = li ? (WHL + (size_t)((li - 1) * 2) * NG * HK) : WH0;
    const float* bf = BIASB + (size_t)(li * 2 + 0) * NG;
    const float* bb = BIASB + (size_t)(li * 2 + 1) * NG;

    gemm_xw_k<<<tilesBlocks, 256, 0, stream>>>(in, Wxf, bf, xWf, Kpad, Kpad / 32);
    gemm_xw_k<<<tilesBlocks, 256, 0, stream>>>(in, Wxb, bb, xWb, Kpad, Kpad / 32);
    lstm_layer_k<<<2, 1024, 0, stream>>>(Wh0, xWf, lens, out, zbuf);
  }

  // 4) head (final bilstm output lives in buf1)
  final_k<<<BB, 160, 0, stream>>>(buf1, fcw, fcb, oww, owb, lens, (float*)d_out);
}
